// E8RHTLinear_8495445311545
// MI455X (gfx1250) — matrix operand order repaired and checked by reading, not compiled, not run on hardware
//
#include <hip/hip_runtime.h>
#include <stdint.h>

typedef __bf16 v16bf __attribute__((ext_vector_type(16)));
typedef float  v8f   __attribute__((ext_vector_type(8)));

#define N_FEAT 8192
#define ROWS   64

// float -> bf16 with round-to-nearest-even
__device__ __forceinline__ unsigned short f2bf(float f) {
  union { float f; uint32_t u; } v; v.f = f;
  uint32_t u = v.u;
  uint32_t r = u + 0x7FFFu + ((u >> 16) & 1u);
  return (unsigned short)(r >> 16);
}

__device__ __forceinline__ uint32_t pack_bf2(float lo, float hi) {
  return (uint32_t)f2bf(lo) | ((uint32_t)f2bf(hi) << 16);
}

// packed bf16 fma: d = a*b + c   (CDNA5 VOP3P V_PK_FMA_BF16)
__device__ __forceinline__ uint32_t pk_fma_bf16(uint32_t a, uint32_t b, uint32_t c) {
  uint32_t d;
  asm("v_pk_fma_bf16 %0, %1, %2, %3" : "=v"(d) : "v"(a), "v"(b), "v"(c));
  return d;
}

// ---------------- Kernel 1: x_rht = FHT(x * SV), emit bf16 ----------------
__global__ __launch_bounds__(256) void fht_fwd_kernel(const float* __restrict__ x,
                                                      const float* __restrict__ SV,
                                                      unsigned short* __restrict__ out) {
  __shared__ float buf[N_FEAT];
  const int row = blockIdx.x;
  const int tid = threadIdx.x;
  const float* xr = x + (size_t)row * N_FEAT;
  for (int i = tid; i < N_FEAT; i += 256) buf[i] = xr[i] * SV[i];
  __syncthreads();
  for (int h = 1; h < N_FEAT; h <<= 1) {
    #pragma unroll
    for (int t = 0; t < (N_FEAT / 2) / 256; ++t) {
      int j = tid + t * 256;
      int i = ((j & ~(h - 1)) << 1) | (j & (h - 1));
      float a = buf[i], b = buf[i + h];
      buf[i]     = a + b;
      buf[i + h] = a - b;
    }
    __syncthreads();
  }
  const float scale = 0x1.6a09e6p-7f; // 1/sqrt(8192)
  unsigned short* orow = out + (size_t)row * N_FEAT;
  for (int i = tid; i < N_FEAT; i += 256) orow[i] = f2bf(buf[i] * scale);
}

// ---- one pipeline step: decompress B fragment, 4 WMMAs against shared B ----
__device__ __forceinline__ void qstep(
    const uint4& c1a, const uint4& c2a, const uint4& c1b, const uint4& c2b,
    uint32_t spair,
    const uint4& a00, const uint4& a01, const uint4& a10, const uint4& a11,
    const uint4& a20, const uint4& a21, const uint4& a30, const uint4& a31,
    v8f& acc0, v8f& acc1, v8f& acc2, v8f& acc3)
{
  union AF { v16bf v; uint4 q[2]; };
  AF b;
  b.q[0].x = pk_fma_bf16(c2a.x, spair, c1a.x);
  b.q[0].y = pk_fma_bf16(c2a.y, spair, c1a.y);
  b.q[0].z = pk_fma_bf16(c2a.z, spair, c1a.z);
  b.q[0].w = pk_fma_bf16(c2a.w, spair, c1a.w);
  b.q[1].x = pk_fma_bf16(c2b.x, spair, c1b.x);
  b.q[1].y = pk_fma_bf16(c2b.y, spair, c1b.y);
  b.q[1].z = pk_fma_bf16(c2b.z, spair, c1b.z);
  b.q[1].w = pk_fma_bf16(c2b.w, spair, c1b.w);
  AF a;
  a.q[0] = a00; a.q[1] = a01;
  acc0 = __builtin_amdgcn_wmma_f32_16x16x32_bf16(false, a.v, false, b.v, (short)0, acc0, false, false);
  a.q[0] = a10; a.q[1] = a11;
  acc1 = __builtin_amdgcn_wmma_f32_16x16x32_bf16(false, a.v, false, b.v, (short)0, acc1, false, false);
  a.q[0] = a20; a.q[1] = a21;
  acc2 = __builtin_amdgcn_wmma_f32_16x16x32_bf16(false, a.v, false, b.v, (short)0, acc2, false, false);
  a.q[0] = a30; a.q[1] = a31;
  acc3 = __builtin_amdgcn_wmma_f32_16x16x32_bf16(false, a.v, false, b.v, (short)0, acc3, false, false);
}

// ---------------- Kernel 2: y_rht = x_rht @ decompress(W)^T ----------------
// Each wave: 64x16 output strip; software-pipelined (indices 2 deep,
// codebook gathers + A fragments 1 deep).
__global__ __launch_bounds__(256) void qgemm_kernel(
    const unsigned short* __restrict__ Abf,   // x_rht bf16 [64][8192]
    const int*   __restrict__ Q1,
    const int*   __restrict__ Q2,
    const float* __restrict__ cb1f,           // [256][8]
    const float* __restrict__ cb2f,           // [256][8]
    const float* __restrict__ resid_p,
    float*       __restrict__ Y)              // y_rht [64][8192]
{
  __shared__ uint4 cb1[256];
  __shared__ uint4 cb2[256];
  const int tid = threadIdx.x;
  { // convert both codebooks to packed bf16 in LDS (one entry each per thread)
    const float* s1 = cb1f + tid * 8;
    uint4 p1;
    p1.x = pack_bf2(s1[0], s1[1]); p1.y = pack_bf2(s1[2], s1[3]);
    p1.z = pack_bf2(s1[4], s1[5]); p1.w = pack_bf2(s1[6], s1[7]);
    cb1[tid] = p1;
    const float* s2 = cb2f + tid * 8;
    uint4 p2;
    p2.x = pack_bf2(s2[0], s2[1]); p2.y = pack_bf2(s2[2], s2[3]);
    p2.z = pack_bf2(s2[4], s2[5]); p2.w = pack_bf2(s2[6], s2[7]);
    cb2[tid] = p2;
  }
  __syncthreads();

  const unsigned short rb = f2bf(resid_p[0]);
  const uint32_t spair = (uint32_t)rb | ((uint32_t)rb << 16);

  const int lane = tid & 31;
  const int wave = tid >> 5;
  const int n0   = (blockIdx.x * 8 + wave) * 16; // output-feature base
  const int lr   = lane & 15;
  const int hi   = lane >> 4;                    // 0: K 0..15, 1: K 16..31 (B frag)
  const int m    = n0 + lr;                      // this lane's weight row (B column)

  // B: lane holds 16 consecutive K of column m -> 2 adjacent 8-groups
  const int2* __restrict__ q1p = (const int2*)(Q1 + (size_t)m * (N_FEAT / 8) + hi * 2);
  const int2* __restrict__ q2p = (const int2*)(Q2 + (size_t)m * (N_FEAT / 8) + hi * 2);
  // A: lane holds 16 consecutive K of row lr (lo lane K k..k+15, hi lane K k+16..k+31)
  const unsigned short* __restrict__ arow = Abf + (size_t)lr * N_FEAT + hi * 16;

  v8f acc0 = {}, acc1 = {}, acc2 = {}, acc3 = {};

  // ---- pipeline prologue ----
  int2 i1c = q1p[0], i2c = q2p[0];   // indices for k=0
  int2 i1n = q1p[2], i2n = q2p[2];   // indices for k=32
  // codebook gathers for k=0
  uint4 c1a = cb1[i1c.x], c2a = cb2[i2c.x];
  uint4 c1b = cb1[i1c.y], c2b = cb2[i2c.y];
  // A fragments for k=0
  uint4 a00 = *(const uint4*)(arow);
  uint4 a01 = *(const uint4*)(arow + 8);
  uint4 a10 = *(const uint4*)(arow + 16 * N_FEAT);
  uint4 a11 = *(const uint4*)(arow + 16 * N_FEAT + 8);
  uint4 a20 = *(const uint4*)(arow + 32 * N_FEAT);
  uint4 a21 = *(const uint4*)(arow + 32 * N_FEAT + 8);
  uint4 a30 = *(const uint4*)(arow + 48 * N_FEAT);
  uint4 a31 = *(const uint4*)(arow + 48 * N_FEAT + 8);

  for (int k = 0; k < N_FEAT - 32; k += 32) {
    // prefetch indices for k+64 (clamped on the tail to stay in bounds)
    const int kf = (k + 64 < N_FEAT) ? (k + 64) : 0;
    int2 i1f = q1p[kf >> 4];
    int2 i2f = q2p[kf >> 4];
    // issue codebook gathers for k+32 (indices loaded one iteration ago)
    uint4 nc1a = cb1[i1n.x], nc2a = cb2[i2n.x];
    uint4 nc1b = cb1[i1n.y], nc2b = cb2[i2n.y];
    // prefetch A fragments for k+32
    const unsigned short* ap = arow + (k + 32);
    uint4 n00 = *(const uint4*)(ap);
    uint4 n01 = *(const uint4*)(ap + 8);
    uint4 n10 = *(const uint4*)(ap + 16 * N_FEAT);
    uint4 n11 = *(const uint4*)(ap + 16 * N_FEAT + 8);
    uint4 n20 = *(const uint4*)(ap + 32 * N_FEAT);
    uint4 n21 = *(const uint4*)(ap + 32 * N_FEAT + 8);
    uint4 n30 = *(const uint4*)(ap + 48 * N_FEAT);
    uint4 n31 = *(const uint4*)(ap + 48 * N_FEAT + 8);

    // compute iteration k (operands all fetched >= 1 iteration ago)
    qstep(c1a, c2a, c1b, c2b, spair,
          a00, a01, a10, a11, a20, a21, a30, a31,
          acc0, acc1, acc2, acc3);

    // rotate pipeline registers
    c1a = nc1a; c2a = nc2a; c1b = nc1b; c2b = nc2b;
    i1n = i1f;  i2n = i2f;
    a00 = n00; a01 = n01; a10 = n10; a11 = n11;
    a20 = n20; a21 = n21; a30 = n30; a31 = n31;
  }
  // ---- epilogue: final iteration (k = N_FEAT - 32) ----
  qstep(c1a, c2a, c1b, c2b, spair,
        a00, a01, a10, a11, a20, a21, a30, a31,
        acc0, acc1, acc2, acc3);

  // C/D layout: VGPR r -> M=r (lanes 0-15) / M=8+r (lanes 16-31), N = lane&15
  const int ncol = n0 + lr;
  const int mb   = hi * 8;
  float* y0 = Y + ncol;
  #pragma unroll
  for (int r = 0; r < 8; ++r) {
    y0[(size_t)(mb + r) * N_FEAT]      = acc0[r];
    y0[(size_t)(16 + mb + r) * N_FEAT] = acc1[r];
    y0[(size_t)(32 + mb + r) * N_FEAT] = acc2[r];
    y0[(size_t)(48 + mb + r) * N_FEAT] = acc3[r];
  }
}

// ---------------- Kernel 3: y = FHT(y_rht) * Wscale * SU ----------------
__global__ __launch_bounds__(256) void fht_out_kernel(const float* __restrict__ yrht,
                                                      const float* __restrict__ SU,
                                                      const float* __restrict__ Wscale_p,
                                                      float* __restrict__ out) {
  __shared__ float buf[N_FEAT];
  const int row = blockIdx.x;
  const int tid = threadIdx.x;
  const float* yr = yrht + (size_t)row * N_FEAT;
  for (int i = tid; i < N_FEAT; i += 256) buf[i] = yr[i];
  __syncthreads();
  for (int h = 1; h < N_FEAT; h <<= 1) {
    #pragma unroll
    for (int t = 0; t < (N_FEAT / 2) / 256; ++t) {
      int j = tid + t * 256;
      int i = ((j & ~(h - 1)) << 1) | (j & (h - 1));
      float a = buf[i], b = buf[i + h];
      buf[i]     = a + b;
      buf[i + h] = a - b;
    }
    __syncthreads();
  }
  const float scale = 0x1.6a09e6p-7f * Wscale_p[0]; // 1/sqrt(8192) * Wscale
  float* orow = out + (size_t)row * N_FEAT;
  for (int i = tid; i < N_FEAT; i += 256) orow[i] = buf[i] * scale * SU[i];
}

extern "C" void kernel_launch(void* const* d_in, const int* in_sizes, int n_in,
                              void* d_out, int out_size, void* d_ws, size_t ws_size,
                              hipStream_t stream) {
  const float* x   = (const float*)d_in[0];
  const int*   Q1  = (const int*)d_in[1];
  const int*   Q2  = (const int*)d_in[2];
  const float* cb1 = (const float*)d_in[3];
  const float* cb2 = (const float*)d_in[4];
  const float* SU  = (const float*)d_in[5];
  const float* SV  = (const float*)d_in[6];
  const float* Wsc = (const float*)d_in[7];
  const float* irs = (const float*)d_in[8];

  unsigned short* xrht = (unsigned short*)d_ws;                                          // 1 MB bf16
  float* yrht = (float*)((char*)d_ws + (size_t)ROWS * N_FEAT * sizeof(unsigned short)); // 2 MB f32

  fht_fwd_kernel<<<ROWS, 256, 0, stream>>>(x, SV, xrht);
  qgemm_kernel<<<N_FEAT / (16 * 8), 256, 0, stream>>>(xrht, Q1, Q2, cb1, cb2, irs, yrht);
  fht_out_kernel<<<ROWS, 256, 0, stream>>>(yrht, SU, Wsc, (float*)d_out);
}
